// ScaledDotProductAttention_up_44272522887183
// MI455X (gfx1250) — matrix operand order repaired and checked by reading, not compiled, not run on hardware
//
#include <hip/hip_runtime.h>

// ---------------------------------------------------------------------------
// Scaled dot-product attention (with >=0 -> -inf masking) for gfx1250 (CDNA5)
// B=2 H=12 S=2048 D=64, fp32 in/out.  Outputs: context [B,H,S,D] then
// attn [B,H,S,S] concatenated in d_out.
//
// bf16 WMMA 16x16x32.  Scores computed TRANSPOSED (S^T = K_tile @ Q^T) so
// each lane owns one query row -> row-sum, normalization and repacking of
// exp'd scores into the P@V A-fragment are lane-local (no LDS/shuffles).
// Two passes over K per query tile (sum, then write+accumulate) to avoid
// re-touching the 402MB attn buffer.
//
//  - K pre-converted to bf16 in workspace: inner loop loads 2x16B per
//    A-fragment (no f32->bf16 cvt in the hot loop).
//  - 1/sqrt(d) * log2(e) folded into the Q fragment (scores emerge in log2
//    domain); pass-2 normalization folded as +log2(recip) in the exponent.
//  - exp via __builtin_amdgcn_exp2f -> single v_exp_f32 (no libm subnormal
//    guard); inputs are range-safe and masked lanes are zeroed anyway.
// Workspace: Vt[bh][d][s] bf16 (6.29MB) + Kb[bh][s][d] bf16 (6.29MB).
// ---------------------------------------------------------------------------

typedef __attribute__((ext_vector_type(16))) __bf16 v16bf;
typedef __attribute__((ext_vector_type(8)))  __bf16 v8bf;
typedef __attribute__((ext_vector_type(8)))  float  v8f;

#define S_LEN 2048
#define D_HEAD 64
#define N_BH 24                      // B*H
#define WAVES_PER_BLOCK 8
#define QROWS_PER_WAVE 16
#define QROWS_PER_BLOCK (WAVES_PER_BLOCK * QROWS_PER_WAVE)   // 128
// 1/sqrt(64) * log2(e): scores come out of WMMA already in log2 domain
#define QFOLD 0.180336880f

__device__ __forceinline__ v8f wmma_bf16(v16bf a, v16bf b, v8f c) {
  return __builtin_amdgcn_wmma_f32_16x16x32_bf16(
      /*neg_a=*/false, a, /*neg_b=*/false, b,
      /*c_mod=*/(short)0, c, /*reuse_a=*/false, /*reuse_b=*/false);
}

// Raw v_exp_f32 (2^x).  Safe: |x| is far from the subnormal-result range.
__device__ __forceinline__ float fast_exp2(float x) {
  return __builtin_amdgcn_exp2f(x);
}

// Load 16 contiguous f32, scale by QFOLD, convert to bf16 fragment
// (B-fragment of the score WMMA: lane n=l&15 is the query row).
__device__ __forceinline__ v16bf load_q_bfrag(const float* __restrict__ p) {
  v16bf r;
#pragma unroll
  for (int j = 0; j < 4; ++j) {
    float4 x = ((const float4*)p)[j];
    r[4 * j + 0] = (__bf16)(x.x * QFOLD);
    r[4 * j + 1] = (__bf16)(x.y * QFOLD);
    r[4 * j + 2] = (__bf16)(x.z * QFOLD);
    r[4 * j + 3] = (__bf16)(x.w * QFOLD);
  }
  return r;
}

// A-fragment of the score WMMA from a pre-converted bf16 K row.
// Contiguous k order (matches the Q B-fragment): element e -> k = 16*g + e.
// Two contiguous 16B loads per fragment.
union AFrag { v16bf v; v8bf h[2]; };
__device__ __forceinline__ v16bf load_k_afrag(const __bf16* __restrict__ krow,
                                              int c, int g) {
  AFrag a;
  a.h[0] = *(const v8bf*)(krow + c * 32 + g * 16);       // k = 16g..16g+7
  a.h[1] = *(const v8bf*)(krow + c * 32 + g * 16 + 8);   // k = 16g+8..+15
  return a.v;
}

// One 16-key score^T tile: C[m=key][n=q] = sum_d Kb[key][d] * Q'[q][d]
__device__ __forceinline__ v8f score_tile(const __bf16* __restrict__ krow,
                                          v16bf bq0, v16bf bq1, int g) {
  v8f c = {};
  c = wmma_bf16(load_k_afrag(krow, 0, g), bq0, c);
  c = wmma_bf16(load_k_afrag(krow, 1, g), bq1, c);
  return c;
}

// ---------------------------------------------------------------------------
// Pre-pass A: V [bh][s][d] f32 -> Vt [bh][d][s] bf16 (LDS-tiled transpose)
// grid (N_BH, S/64), block 256
// ---------------------------------------------------------------------------
__global__ void vt_transpose_kernel(const float* __restrict__ V,
                                    __bf16* __restrict__ Vt) {
  __shared__ float tile[64][65];
  const int bh = blockIdx.x;
  const int s0 = blockIdx.y * 64;
  const int t = threadIdx.x;
  const float* Vp = V + ((size_t)bh * S_LEN + s0) * D_HEAD;
#pragma unroll
  for (int i = 0; i < 16; ++i) {
    int r = (t >> 6) + i * 4;   // s within tile
    int c = t & 63;             // d
    tile[r][c] = Vp[(size_t)r * D_HEAD + c];
  }
  __syncthreads();
  __bf16* Vtp = Vt + (size_t)bh * D_HEAD * S_LEN + s0;
#pragma unroll
  for (int i = 0; i < 16; ++i) {
    int d = (t >> 6) + i * 4;
    int s = t & 63;
    Vtp[(size_t)d * S_LEN + s] = (__bf16)tile[s][d];
  }
}

// ---------------------------------------------------------------------------
// Pre-pass B: K f32 -> Kb bf16 (elementwise, coalesced; 4 elems/thread)
// ---------------------------------------------------------------------------
__global__ void k_bf16_kernel(const float* __restrict__ K,
                              __bf16* __restrict__ Kb) {
  size_t i = ((size_t)blockIdx.x * blockDim.x + threadIdx.x) * 4;
  float4 x = *(const float4*)(K + i);
  __bf16 y[4] = {(__bf16)x.x, (__bf16)x.y, (__bf16)x.z, (__bf16)x.w};
  *(ulong1*)(Kb + i) = *(ulong1*)y;   // 8B store of 4 bf16
}

// ---------------------------------------------------------------------------
// Main attention kernel.  grid (N_BH, S/128), block 256 (8 waves).
// Each wave owns 16 query rows.
// ---------------------------------------------------------------------------
__global__ void __launch_bounds__(256)
attn_kernel(const float* __restrict__ Q, const __bf16* __restrict__ Kb,
            const __bf16* __restrict__ Vt, float* __restrict__ ctx_out,
            float* __restrict__ attn_out) {
  const int bh   = blockIdx.x;
  const int wave = threadIdx.x >> 5;
  const int lane = threadIdx.x & 31;
  const int g    = lane >> 4;     // lane group (key-half selector)
  const int ln   = lane & 15;     // this lane's query row within the tile
  const int qbase = blockIdx.y * QROWS_PER_BLOCK + wave * QROWS_PER_WAVE;

  const __bf16* Kbh = Kb + (size_t)bh * S_LEN * D_HEAD;
  const __bf16* Vbh = Vt + (size_t)bh * D_HEAD * S_LEN;

  // Persistent Q B-fragments, pre-scaled by 1/sqrt(d)*log2(e)
  const float* Qrow = Q + ((size_t)bh * S_LEN + (qbase + ln)) * D_HEAD;
  const v16bf bq0 = load_q_bfrag(Qrow + g * 16);
  const v16bf bq1 = load_q_bfrag(Qrow + 32 + g * 16);

  // ---------------- Pass 1: row sums of masked exp ----------------
  // score s is in log2 domain; mask: s >= 0 -> 0, else exp2(s)
  float sum = 0.0f;
  for (int tile = 0; tile < S_LEN / 16; ++tile) {
    const __bf16* krow = Kbh + (size_t)(tile * 16 + ln) * D_HEAD;
    v8f s = score_tile(krow, bq0, bq1, g);
#pragma unroll
    for (int r = 0; r < 8; ++r) {
      float e = fast_exp2(s[r]);
      sum += (s[r] < 0.0f) ? e : 0.0f;
    }
  }
  sum += __shfl_xor(sum, 16, 32);              // merge the two key halves
  // normalization folded into the exponent: e/sum_total = exp2(s + lr)
  const float lr = -__log2f(sum + 1e-8f);

  // ---------------- Pass 2: write attn, accumulate context ----------------
  v8f ctx[4];
#pragma unroll
  for (int dt = 0; dt < 4; ++dt) { v8f z = {}; ctx[dt] = z; }

  float* arow = attn_out + ((size_t)bh * S_LEN + (qbase + ln)) * S_LEN;

  for (int ch = 0; ch < S_LEN / 32; ++ch) {
    const int k0 = ch * 32;
    v8f s0 = score_tile(Kbh + (size_t)(k0 + ln) * D_HEAD,      bq0, bq1, g);
    v8f s1 = score_tile(Kbh + (size_t)(k0 + 16 + ln) * D_HEAD, bq0, bq1, g);

    // normalized masked exp, pack into P A-fragment (lane-local!)
    v16bf ap;
    float e0[8], e1[8];
#pragma unroll
    for (int r = 0; r < 8; ++r) {
      float a = fast_exp2(s0[r] + lr);
      float b = fast_exp2(s1[r] + lr);
      a = (s0[r] < 0.0f) ? a : 0.0f;
      b = (s1[r] < 0.0f) ? b : 0.0f;
      e0[r] = a;  e1[r] = b;
      ap[r]     = (__bf16)a;   // A elem e<8  -> key = 8g + e    (tile 0)
      ap[8 + r] = (__bf16)b;   // A elem e>=8 -> key = 16+8g+e-8 (tile 1)
    }

    // store normalized attention (lane q = qbase+ln, keys k0+8g.. / +16+8g..)
    float4* d0 = (float4*)(arow + k0 + g * 8);
    d0[0] = make_float4(e0[0], e0[1], e0[2], e0[3]);
    d0[1] = make_float4(e0[4], e0[5], e0[6], e0[7]);
    float4* d1 = (float4*)(arow + k0 + 16 + g * 8);
    d1[0] = make_float4(e1[0], e1[1], e1[2], e1[3]);
    d1[1] = make_float4(e1[4], e1[5], e1[6], e1[7]);

    // context += P(16q x 32k) @ V(32k x 64d), 4 d-tiles of 16
#pragma unroll
    for (int dt = 0; dt < 4; ++dt) {
      const __bf16* vp = Vbh + (size_t)(dt * 16 + ln) * S_LEN + k0 + g * 8;
      AFrag bv;   // keys 8g..8g+7 and 16+8g..16+8g+7 (2x16B, aligned)
      bv.h[0] = *(const v8bf*)vp;
      bv.h[1] = *(const v8bf*)(vp + 16);
      ctx[dt] = wmma_bf16(ap, bv.v, ctx[dt]);
    }
  }

  // write context: C layout lane -> d = dt*16+ln, rows m = 8g + r
  float* obase = ctx_out + ((size_t)bh * S_LEN + qbase) * D_HEAD;
#pragma unroll
  for (int dt = 0; dt < 4; ++dt) {
#pragma unroll
    for (int r = 0; r < 8; ++r) {
      obase[(size_t)(g * 8 + r) * D_HEAD + dt * 16 + ln] = ctx[dt][r];
    }
  }
}

// ---------------------------------------------------------------------------
extern "C" void kernel_launch(void* const* d_in, const int* in_sizes, int n_in,
                              void* d_out, int out_size, void* d_ws,
                              size_t ws_size, hipStream_t stream) {
  const float* Q = (const float*)d_in[0];
  const float* K = (const float*)d_in[1];
  const float* V = (const float*)d_in[2];

  float* ctx  = (float*)d_out;                                   // [B,H,S,D]
  float* attn = ctx + (size_t)N_BH * S_LEN * D_HEAD;             // [B,H,S,S]

  const size_t elems = (size_t)N_BH * S_LEN * D_HEAD;            // 3,145,728
  __bf16* Vt = (__bf16*)d_ws;            // [bh][d][s] bf16, 6.29MB
  __bf16* Kb = Vt + elems;               // [bh][s][d] bf16, 6.29MB

  vt_transpose_kernel<<<dim3(N_BH, S_LEN / 64), 256, 0, stream>>>(V, Vt);
  k_bf16_kernel<<<elems / (256 * 4), 256, 0, stream>>>(K, Kb);
  attn_kernel<<<dim3(N_BH, S_LEN / QROWS_PER_BLOCK),
                WAVES_PER_BLOCK * 32, 0, stream>>>(Q, Kb, Vt, ctx, attn);
}
